// SigmoidLookups_56719338111468
// MI455X (gfx1250) — compile-verified, matrix operand order checked by reading, not run on hardware
//
#include <hip/hip_runtime.h>
#include <hip/hip_bf16.h>

typedef __attribute__((ext_vector_type(16))) __bf16        v16bf;
typedef __attribute__((ext_vector_type(8)))  float         v8f;
typedef __attribute__((ext_vector_type(8)))  unsigned int  v8u;

#define B_   2
#define L_   4096
#define D_   512
#define H_   8
#define HD_  64
#define KK_  64
#define ND_  1536

// ---- workspace layout (bytes) ----
static constexpr size_t OFF_WBQKV = 0;                                   // 1536*512 bf16
static constexpr size_t OFF_WBOUT = OFF_WBQKV + (size_t)1536 * 512 * 2;  // 512*512 bf16
static constexpr size_t OFF_SEL   = OFF_WBOUT + (size_t)512 * 512 * 2;   // B*L fp32
static constexpr size_t OFF_IDX   = OFF_SEL + (size_t)B_ * L_ * 4;       // B*64 int
static constexpr size_t OFF_QB    = OFF_IDX + 1024;                      // [B][H][L][64] bf16
static constexpr size_t OFF_KB    = OFF_QB + (size_t)B_ * H_ * L_ * HD_ * 2;
static constexpr size_t OFF_VT    = OFF_KB + (size_t)B_ * H_ * L_ * HD_ * 2; // [B][H][64][L]
static constexpr size_t OFF_CTX   = OFF_VT + (size_t)B_ * H_ * HD_ * L_ * 2; // [B][64][512] bf16
static constexpr size_t OFF_XB    = OFF_CTX + (size_t)B_ * KK_ * D_ * 2;     // x in bf16

__device__ __forceinline__ unsigned short f2bf(float f) {
  unsigned int u = __builtin_bit_cast(unsigned int, f);
  u += 0x7FFFu + ((u >> 16) & 1u);          // round-to-nearest-even
  return (unsigned short)(u >> 16);
}

__device__ __forceinline__ v8f wmma_bf16(v8u a, v8u b, v8f c) {
  return __builtin_amdgcn_wmma_f32_16x16x32_bf16(
      false, __builtin_bit_cast(v16bf, a),
      false, __builtin_bit_cast(v16bf, b),
      (short)0, c, false, false);
}

// dword index map inside a 32-element (16-dword) bf16 K-chunk for A/B fragments
__device__ __forceinline__ void kdmap(int half, int kd[8]) {
#pragma unroll
  for (int p = 0; p < 8; ++p) kd[p] = p + 4 * half + ((p >= 4) ? 4 : 0);
}

// ---------------- fp32 -> bf16 conversion ----------------
__global__ void convert_kernel(const float* __restrict__ s, unsigned short* __restrict__ d, int n) {
  for (int i = blockIdx.x * blockDim.x + threadIdx.x; i < n; i += gridDim.x * blockDim.x)
    d[i] = f2bf(s[i]);
}

// ---------------- y = x ----------------
__global__ void copy_kernel(const float4* __restrict__ src, float4* __restrict__ dst, int n) {
  int i = blockIdx.x * blockDim.x + threadIdx.x;
  if (i < n) dst[i] = src[i];
}

// ---------------- selection = sigmoid(x @ sel_w) ----------------
__global__ __launch_bounds__(256) void sel_kernel(const float* __restrict__ x,
                                                  const float* __restrict__ sel_w,
                                                  float* __restrict__ sel) {
  int row  = blockIdx.x * 8 + (threadIdx.x >> 5);
  int lane = threadIdx.x & 31;
  const float* xr = x + (size_t)row * D_;
  float s = 0.f;
#pragma unroll
  for (int i = 0; i < 16; ++i) s += xr[lane + i * 32] * sel_w[lane + i * 32];
#pragma unroll
  for (int m = 16; m > 0; m >>= 1) s += __shfl_xor(s, m, 32);
  if (lane == 0) sel[row] = 1.f / (1.f + __expf(-s));
}

// ---------------- per-batch top-64 (iterative argmax) ----------------
__global__ __launch_bounds__(256) void topk_kernel(const float* __restrict__ sel, int* __restrict__ idx) {
  __shared__ float vals[L_];
  __shared__ float rv[256];
  __shared__ int   ri[256];
  int b = blockIdx.x, tid = threadIdx.x;
  for (int i = tid; i < L_; i += 256) vals[i] = sel[(size_t)b * L_ + i];
  __syncthreads();
  for (int round = 0; round < KK_; ++round) {
    float best = -1e30f; int bi = 0;
    for (int i = tid; i < L_; i += 256) {
      float v = vals[i];
      if (v > best) { best = v; bi = i; }
    }
    rv[tid] = best; ri[tid] = bi;
    __syncthreads();
    for (int s = 128; s > 0; s >>= 1) {
      if (tid < s && rv[tid + s] > rv[tid]) { rv[tid] = rv[tid + s]; ri[tid] = ri[tid + s]; }
      __syncthreads();
    }
    if (tid == 0) { idx[b * KK_ + round] = ri[0]; vals[ri[0]] = -1e30f; }
    __syncthreads();
  }
}

// ---------------- QKV projection: (8192,512) x (1536,512)^T, bf16 WMMA ----------------
// Double-buffered async global->LDS staging (ASYNCcnt), LDS row pitch 80B:
// 16B-aligned for b128 async writes, 20-dword stride = conflict-free fragment reads.
#define QKV_PITCH 40  // u16 per LDS row

__global__ __launch_bounds__(256) void qkv_kernel(const unsigned short* __restrict__ xb,
                                                  const unsigned short* __restrict__ Wb,
                                                  const float* __restrict__ Wqkv_b,
                                                  unsigned short* __restrict__ Qb,
                                                  unsigned short* __restrict__ Kb,
                                                  unsigned short* __restrict__ Vt) {
  __shared__ unsigned short As[2][128 * QKV_PITCH];
  __shared__ unsigned short Bs[2][128 * QKV_PITCH];
  const int tid = threadIdx.x;
  const int lane = tid & 31, half = lane >> 4, l16 = lane & 15;
  const int wave = tid >> 5, waveM = wave >> 2, waveN = wave & 3;
  int kd[8]; kdmap(half, kd);
  const int mBase = blockIdx.y * 128;
  const int nBase = blockIdx.x * 128;

  // staging assignment: 2 threads per tile-row, each copies 32B (2 x b128)
  const int srow = tid >> 1;
  const int scol = (tid & 1) * 16;  // u16 offset within 32-elem tile row
  const unsigned short* gA = xb + (size_t)(mBase + srow) * D_ + scol;
  const unsigned short* gB = Wb + (size_t)(nBase + srow) * D_ + scol;
  const unsigned ldsA0 = (unsigned)(uintptr_t)&As[0][srow * QKV_PITCH + scol];
  const unsigned ldsB0 = (unsigned)(uintptr_t)&Bs[0][srow * QKV_PITCH + scol];
  const unsigned bufStride = (unsigned)(128 * QKV_PITCH * 2);

  auto issue = [&](int kt) {
    const int buf = kt & 1;
    unsigned long long a0 = (unsigned long long)(uintptr_t)(gA + kt * 32);
    unsigned long long b0 = (unsigned long long)(uintptr_t)(gB + kt * 32);
    unsigned la = ldsA0 + buf * bufStride;
    unsigned lb = ldsB0 + buf * bufStride;
    // inst offset applies to BOTH lds and global address (ISA 08 §4.4)
    asm volatile("global_load_async_to_lds_b128 %0, %1, off" :: "v"(la), "v"(a0) : "memory");
    asm volatile("global_load_async_to_lds_b128 %0, %1, off offset:16" :: "v"(la), "v"(a0) : "memory");
    asm volatile("global_load_async_to_lds_b128 %0, %1, off" :: "v"(lb), "v"(b0) : "memory");
    asm volatile("global_load_async_to_lds_b128 %0, %1, off offset:16" :: "v"(lb), "v"(b0) : "memory");
  };

  v8f acc[4][2] = {};
  issue(0);
  for (int kt = 0; kt < 16; ++kt) {
    if (kt + 1 < 16) {
      issue(kt + 1);  // prefetch next tile into the other buffer
      asm volatile("s_wait_asynccnt 0x4" ::: "memory");  // in-order: first 4 done
    } else {
      asm volatile("s_wait_asynccnt 0x0" ::: "memory");
    }
    __syncthreads();
    const int buf = kt & 1;
    v8u af[4], bfr[2];
#pragma unroll
    for (int i = 0; i < 4; ++i) {
      const unsigned int* ar = (const unsigned int*)&As[buf][(waveM * 64 + i * 16 + l16) * QKV_PITCH];
#pragma unroll
      for (int p = 0; p < 8; ++p) af[i][p] = ar[kd[p]];
    }
#pragma unroll
    for (int j = 0; j < 2; ++j) {
      const unsigned int* br = (const unsigned int*)&Bs[buf][(waveN * 32 + j * 16 + l16) * QKV_PITCH];
#pragma unroll
      for (int p = 0; p < 8; ++p) bfr[j][p] = br[kd[p]];
    }
#pragma unroll
    for (int i = 0; i < 4; ++i)
#pragma unroll
      for (int j = 0; j < 2; ++j) acc[i][j] = wmma_bf16(af[i], bfr[j], acc[i][j]);
    __syncthreads();
  }

  // epilogue: bias, K-scale, scatter into Q/K (head-major) and V (transposed)
#pragma unroll
  for (int i = 0; i < 4; ++i) {
#pragma unroll
    for (int j = 0; j < 2; ++j) {
      int c = nBase + waveN * 32 + j * 16 + l16;
      int mat = c >> 9, w = c & 511, hh = w >> 6, hd = w & 63;
      float bias = Wqkv_b[c];
#pragma unroll
      for (int r = 0; r < 8; ++r) {
        int m = mBase + waveM * 64 + i * 16 + r + 8 * half;
        int bb = m >> 12, ll = m & 4095;
        float v = acc[i][j][r] + bias;
        if (mat == 1) v *= 0.125f;  // 1/sqrt(HD)
        unsigned short bv = f2bf(v);
        size_t hb = (size_t)(bb * H_ + hh);
        if (mat == 0)      Qb[(hb * L_ + ll) * HD_ + hd] = bv;
        else if (mat == 1) Kb[(hb * L_ + ll) * HD_ + hd] = bv;
        else               Vt[(hb * HD_ + hd) * L_ + ll] = bv;
      }
    }
  }
}

// ---------------- causal flash attention, only the 64 selected rows per batch ----------------
__global__ __launch_bounds__(128) void attn_kernel(const unsigned short* __restrict__ Qb,
                                                   const unsigned short* __restrict__ Kb,
                                                   const unsigned short* __restrict__ Vt,
                                                   const int* __restrict__ idx,
                                                   unsigned short* __restrict__ ctxb) {
  __shared__ unsigned short Pl[4][16][36];  // per-wave P transpose staging
  const int bh = blockIdx.x;
  const int b = bh >> 3, h = bh & 7;
  const int tid = threadIdx.x, lane = tid & 31, half = lane >> 4, l16 = lane & 15;
  const int wave = tid >> 5;
  int kd[8]; kdmap(half, kd);

  const int slot_l = wave * 16 + l16;
  const int t_l = idx[b * KK_ + slot_l];

  const unsigned int* qr = (const unsigned int*)(Qb + ((size_t)(b * H_ + h) * L_ + t_l) * HD_);
  v8u qf[2];
#pragma unroll
  for (int kc = 0; kc < 2; ++kc)
#pragma unroll
    for (int p = 0; p < 8; ++p) qf[kc][p] = qr[kc * 16 + kd[p]];

  int tmax = t_l;
#pragma unroll
  for (int m = 1; m < 32; m <<= 1) tmax = max(tmax, __shfl_xor(tmax, m, 32));
  int tr[8];
#pragma unroll
  for (int r = 0; r < 8; ++r) tr[r] = __shfl(t_l, r + 8 * half, 32);

  float mrow[8], lsum[8];
#pragma unroll
  for (int r = 0; r < 8; ++r) { mrow[r] = -1e30f; lsum[r] = 0.f; }
  v8f O[4] = {};

  const unsigned short* Kbase = Kb + (size_t)(b * H_ + h) * L_ * HD_;
  const unsigned short* Vbase = Vt + (size_t)(b * H_ + h) * HD_ * L_;

  for (int s0 = 0; s0 <= tmax; s0 += 32) {
    v8u kf[2][2];
#pragma unroll
    for (int sub = 0; sub < 2; ++sub) {
      const unsigned int* kr = (const unsigned int*)(Kbase + (size_t)(s0 + sub * 16 + l16) * HD_);
#pragma unroll
      for (int kc = 0; kc < 2; ++kc)
#pragma unroll
        for (int p = 0; p < 8; ++p) kf[sub][kc][p] = kr[kc * 16 + kd[p]];
    }
    v8f S0 = {}, S1 = {};
    S0 = wmma_bf16(qf[0], kf[0][0], S0);
    S0 = wmma_bf16(qf[1], kf[0][1], S0);
    S1 = wmma_bf16(qf[0], kf[1][0], S1);
    S1 = wmma_bf16(qf[1], kf[1][1], S1);

    float sv0[8], sv1[8], rmax[8];
    const int c0 = s0 + l16, c1 = s0 + 16 + l16;
#pragma unroll
    for (int r = 0; r < 8; ++r) {
      sv0[r] = S0[r] + ((c0 <= tr[r]) ? 0.f : -10000.f);
      sv1[r] = S1[r] + ((c1 <= tr[r]) ? 0.f : -10000.f);
      rmax[r] = fmaxf(sv0[r], sv1[r]);
    }
#pragma unroll
    for (int mm = 1; mm < 16; mm <<= 1)
#pragma unroll
      for (int r = 0; r < 8; ++r) rmax[r] = fmaxf(rmax[r], __shfl_xor(rmax[r], mm, 32));

    float fac[8], rsum[8];
#pragma unroll
    for (int r = 0; r < 8; ++r) {
      float mn = fmaxf(mrow[r], rmax[r]);
      fac[r] = __expf(mrow[r] - mn);
      sv0[r] = __expf(sv0[r] - mn);
      sv1[r] = __expf(sv1[r] - mn);
      mrow[r] = mn;
      rsum[r] = sv0[r] + sv1[r];
    }
#pragma unroll
    for (int mm = 1; mm < 16; mm <<= 1)
#pragma unroll
      for (int r = 0; r < 8; ++r) rsum[r] += __shfl_xor(rsum[r], mm, 32);
#pragma unroll
    for (int r = 0; r < 8; ++r) lsum[r] = lsum[r] * fac[r] + rsum[r];
#pragma unroll
    for (int c = 0; c < 4; ++c)
#pragma unroll
      for (int r = 0; r < 8; ++r) O[c][r] *= fac[r];

    // transpose P (C-layout) -> A-fragment layout via LDS (wave-private tile)
#pragma unroll
    for (int r = 0; r < 8; ++r) {
      Pl[wave][r + 8 * half][l16]      = f2bf(sv0[r]);
      Pl[wave][r + 8 * half][16 + l16] = f2bf(sv1[r]);
    }
    asm volatile("s_wait_dscnt 0" ::: "memory");
    v8u pf;
    const unsigned int* pr = (const unsigned int*)&Pl[wave][l16][0];
#pragma unroll
    for (int p = 0; p < 8; ++p) pf[p] = pr[kd[p]];

#pragma unroll
    for (int c = 0; c < 4; ++c) {
      const unsigned int* vr = (const unsigned int*)(Vbase + (size_t)(c * 16 + l16) * L_);
      v8u vf;
#pragma unroll
      for (int p = 0; p < 8; ++p) vf[p] = vr[(s0 >> 1) + kd[p]];
      O[c] = wmma_bf16(pf, vf, O[c]);
    }
    asm volatile("s_wait_dscnt 0" ::: "memory");
  }

  float inv[8];
#pragma unroll
  for (int r = 0; r < 8; ++r) inv[r] = 1.f / lsum[r];
#pragma unroll
  for (int c = 0; c < 4; ++c)
#pragma unroll
    for (int r = 0; r < 8; ++r) {
      int slot = wave * 16 + r + 8 * half;
      ctxb[((size_t)b * KK_ + slot) * D_ + h * HD_ + c * 16 + l16] = f2bf(O[c][r] * inv[r]);
    }
}

// ---------------- out-proj (128 rows x 512) + gate + scatter-add into y ----------------
__global__ __launch_bounds__(128) void outproj_kernel(const unsigned short* __restrict__ ctxb,
                                                      const unsigned short* __restrict__ Wob,
                                                      const float* __restrict__ out_b,
                                                      const float* __restrict__ sel,
                                                      const int* __restrict__ idx,
                                                      float* __restrict__ y) {
  const int b = blockIdx.y;
  const int colBase = blockIdx.x * 64;
  const int tid = threadIdx.x, lane = tid & 31, half = lane >> 4, l16 = lane & 15;
  const int wave = tid >> 5;
  int kd[8]; kdmap(half, kd);

  const int slot_l = wave * 16 + l16;
  const unsigned int* ar = (const unsigned int*)(ctxb + ((size_t)b * KK_ + slot_l) * D_);
  v8f acc[4] = {};

  for (int kk = 0; kk < D_; kk += 32) {
    v8u af;
#pragma unroll
    for (int p = 0; p < 8; ++p) af[p] = ar[(kk >> 1) + kd[p]];
#pragma unroll
    for (int c = 0; c < 4; ++c) {
      const unsigned int* br = (const unsigned int*)(Wob + (size_t)(colBase + c * 16 + l16) * D_);
      v8u bw;
#pragma unroll
      for (int p = 0; p < 8; ++p) bw[p] = br[(kk >> 1) + kd[p]];
      acc[c] = wmma_bf16(af, bw, acc[c]);
    }
  }
#pragma unroll
  for (int c = 0; c < 4; ++c) {
    int col = colBase + c * 16 + l16;
    float ob = out_b[col];
#pragma unroll
    for (int r = 0; r < 8; ++r) {
      int slot = wave * 16 + r + 8 * half;
      int t = idx[b * KK_ + slot];
      float g = (acc[c][r] + ob) * sel[(size_t)b * L_ + t];
      y[((size_t)b * L_ + t) * D_ + col] += g;  // unique (t,col) per lane-element
    }
  }
}

extern "C" void kernel_launch(void* const* d_in, const int* in_sizes, int n_in,
                              void* d_out, int out_size, void* d_ws, size_t ws_size,
                              hipStream_t stream) {
  (void)in_sizes; (void)n_in; (void)out_size; (void)ws_size;
  const float* x      = (const float*)d_in[0];
  const float* Wqkv_w = (const float*)d_in[1];
  const float* Wqkv_b = (const float*)d_in[2];
  const float* sel_w  = (const float*)d_in[3];
  const float* out_w  = (const float*)d_in[4];
  const float* out_b  = (const float*)d_in[5];
  float* y = (float*)d_out;
  char* ws = (char*)d_ws;

  unsigned short* Wbqkv = (unsigned short*)(ws + OFF_WBQKV);
  unsigned short* Wbout = (unsigned short*)(ws + OFF_WBOUT);
  float*          sel   = (float*)(ws + OFF_SEL);
  int*            idx   = (int*)(ws + OFF_IDX);
  unsigned short* Qb    = (unsigned short*)(ws + OFF_QB);
  unsigned short* Kb    = (unsigned short*)(ws + OFF_KB);
  unsigned short* Vt    = (unsigned short*)(ws + OFF_VT);
  unsigned short* ctxb  = (unsigned short*)(ws + OFF_CTX);
  unsigned short* xb    = (unsigned short*)(ws + OFF_XB);

  convert_kernel<<<768, 256, 0, stream>>>(Wqkv_w, Wbqkv, ND_ * D_);
  convert_kernel<<<256, 256, 0, stream>>>(out_w, Wbout, D_ * D_);
  convert_kernel<<<2048, 256, 0, stream>>>(x, xb, B_ * L_ * D_);
  copy_kernel<<<(B_ * L_ * D_ / 4 + 255) / 256, 256, 0, stream>>>((const float4*)x, (float4*)y,
                                                                  B_ * L_ * D_ / 4);
  sel_kernel<<<(B_ * L_) / 8, 256, 0, stream>>>(x, sel_w, sel);
  topk_kernel<<<B_, 256, 0, stream>>>(sel, idx);
  qkv_kernel<<<dim3(ND_ / 128, (B_ * L_) / 128), 256, 0, stream>>>(xb, Wbqkv, Wqkv_b, Qb, Kb, Vt);
  attn_kernel<<<B_ * H_, 128, 0, stream>>>(Qb, Kb, Vt, idx, ctxb);
  outproj_kernel<<<dim3(D_ / 64, B_), 128, 0, stream>>>(ctxb, Wbout, out_b, sel, idx, y);
}
